// ConvSA_57775900066251
// MI455X (gfx1250) — matrix operand order repaired and checked by reading, not compiled, not run on hardware
//
#include <hip/hip_runtime.h>
#include <hip/hip_bf16.h>
#include <stdint.h>

// ---------------------------------------------------------------------------
// ConvSA: q/k/v = conv3x3(feat), att = softmax(q^T k), out = v*att^T + v
// B=8, C=256, H=W=48 (N=2304), E=512.  GEMMs on v_wmma_f32_16x16x32_bf16,
// operand staging via GLOBAL_LOAD_ASYNC_TO_LDS_B128 (double-buffered in conv).
// ---------------------------------------------------------------------------

#define CIN   256
#define EMB   512
#define HW    48
#define NSP   2304            // 48*48
#define KCONV 2304            // CIN*9

typedef __attribute__((ext_vector_type(16))) __bf16 v16bf;
typedef __attribute__((ext_vector_type(8)))  float  v8f;
typedef __attribute__((ext_vector_type(8)))  unsigned int v8u;
typedef int v4i_vs __attribute__((vector_size(16)));   // matches builtin param

__device__ __forceinline__ unsigned short f2bf(float f) {
    unsigned u = __builtin_bit_cast(unsigned, f);
    unsigned r = u + 0x7FFFu + ((u >> 16) & 1u);   // round-to-nearest-even
    return (unsigned short)(r >> 16);
}

__device__ __forceinline__ v16bf make_frag(uint4 lo, uint4 hi) {
    v8u u;
    u[0] = lo.x; u[1] = lo.y; u[2] = lo.z; u[3] = lo.w;
    u[4] = hi.x; u[5] = hi.y; u[6] = hi.z; u[7] = hi.w;
    return __builtin_bit_cast(v16bf, u);
}

__device__ __forceinline__ v8f wmma_bf16(v16bf a, v16bf b, v8f c) {
    return __builtin_amdgcn_wmma_f32_16x16x32_bf16(
        false, a, false, b, (short)0, c, false, false);
}

// ---- async global->LDS copy (CDNA5 ASYNCcnt path), safe fallback ----------
__device__ __forceinline__ void async_copy_b128(const void* g, void* l) {
#if __has_builtin(__builtin_amdgcn_global_load_async_to_lds_b128)
    __builtin_amdgcn_global_load_async_to_lds_b128(
        (__attribute__((address_space(1))) v4i_vs*)(uintptr_t)g,
        (__attribute__((address_space(3))) v4i_vs*)(unsigned int)(uintptr_t)l,
        0, 0);
#else
    *(uint4*)l = *(const uint4*)g;
#endif
}

__device__ __forceinline__ void wait_async() {
#if __has_builtin(__builtin_amdgcn_global_load_async_to_lds_b128)
#if __has_builtin(__builtin_amdgcn_s_wait_asynccnt)
    __builtin_amdgcn_s_wait_asynccnt(0);
#else
    asm volatile("s_wait_asynccnt 0x0" ::: "memory");
#endif
#endif
}

// ---------------------------------------------------------------------------
// Kernel 1: repack conv weights [O][C][3][3] fp32 -> bf16 [3][EMB][KCONV],
// K ordered as kk = (ry*3+rx)*256 + c.
// ---------------------------------------------------------------------------
__global__ __launch_bounds__(256)
void repack_w_kernel(const float* __restrict__ Wq,
                     const float* __restrict__ Wk,
                     const float* __restrict__ Wv,
                     unsigned short* __restrict__ Wmat) {
    int idx  = blockIdx.x * 256 + threadIdx.x;
    int conv = idx / (EMB * KCONV);
    int rem  = idx % (EMB * KCONV);
    int e    = rem / KCONV;
    int kk   = rem % KCONV;
    int c    = kk & 255;
    int rs   = kk >> 8;
    const float* W = (conv == 0) ? Wq : (conv == 1) ? Wk : Wv;
    float v = W[(((size_t)e * CIN + c) * 3 + rs / 3) * 3 + (rs % 3)];
    Wmat[idx] = f2bf(v);
}

// ---------------------------------------------------------------------------
// Kernel 1b: feat [B][C][H][W] fp32 -> channels-last bf16 featT [B][H][W][C].
// ---------------------------------------------------------------------------
__global__ __launch_bounds__(256)
void featT_kernel(const float* __restrict__ feat,
                  unsigned short* __restrict__ featT) {
    size_t idx = (size_t)blockIdx.x * 256 + threadIdx.x;
    int c = (int)(idx & 255);
    size_t rem = idx >> 8;                 // b*2304 + y*48 + x
    int x = (int)(rem % HW);
    size_t rem2 = rem / HW;
    int y = (int)(rem2 % HW);
    int b = (int)(rem2 / HW);
    featT[idx] = f2bf(feat[(((size_t)b * CIN + c) * HW + y) * HW + x]);
}

// ---------------------------------------------------------------------------
// Kernel 2: conv3x3 as implicit GEMM.  Grid (N/64, E/64, B*3), block 256.
// Block computes 64(E) x 64(N); wave w: E-tile w>>1, N-tiles {2*(w&1),+1}.
// Double-buffered LDS tiles: async DMA of chunk t+1 overlaps WMMA of chunk t.
// ---------------------------------------------------------------------------
__global__ __launch_bounds__(256)
void conv_qkv_wmma(const unsigned short* __restrict__ featT,
                   const unsigned short* __restrict__ Wmat,
                   const float* __restrict__ bq,
                   const float* __restrict__ bk,
                   const float* __restrict__ bv,
                   unsigned short* __restrict__ qT,
                   unsigned short* __restrict__ kT,
                   unsigned short* __restrict__ vB,
                   float* __restrict__ vF) {
    __shared__ unsigned short Al[2][64 * 40];   // weights 64e x 32k (x2 bufs)
    __shared__ unsigned short Bl[2][64 * 40];   // patches 64n x 32k (x2 bufs)

    const int tid  = threadIdx.x;
    const int wave = tid >> 5;
    const int lane = tid & 31;
    const int m16  = lane & 15;
    const int hi   = lane >> 4;

    const int b    = blockIdx.z / 3;
    const int conv = blockIdx.z % 3;
    const int e0   = blockIdx.y * 64;
    const int i0   = blockIdx.x * 64;

    const int et = wave >> 1;          // 0..3  (E sub-tile)
    const int np = (wave & 1) * 2;     // 0 or 2 (first N sub-tile)

    const unsigned short* Wc = Wmat + (size_t)conv * EMB * KCONV;

    const int sm  = tid >> 2;          // 0..63 : staging row (e or n)
    const int seg = tid & 3;           // 16B segment within 64B row chunk

    const int i_n = i0 + sm;
    const int y   = i_n / HW, x = i_n % HW;

    // issue async copies for K-chunk t into LDS buffer buf
    auto stage = [&](int t, int buf) {
        const int rs = t >> 3, cc = t & 7;
        const int kk0 = rs * 256 + cc * 32;
        async_copy_b128(Wc + (size_t)(e0 + sm) * KCONV + kk0 + seg * 8,
                        &Al[buf][sm * 40 + seg * 8]);
        const int yy = y + (rs / 3 - 1), xx = x + (rs % 3 - 1);
        unsigned short* dst = &Bl[buf][sm * 40 + seg * 8];
        if ((yy >= 0) && (yy < HW) && (xx >= 0) && (xx < HW)) {
            async_copy_b128(
                featT + ((((size_t)b * HW + yy) * HW + xx) * CIN + cc * 32 + seg * 8),
                dst);
        } else {
            uint4 z = {0u, 0u, 0u, 0u};
            *(uint4*)dst = z;
        }
    };

    v8f acc[2];
    acc[0] = v8f{}; acc[1] = v8f{};

    stage(0, 0);
    for (int t = 0; t < 72; ++t) {
        const int buf = t & 1;
        wait_async();          // chunk t's DMA done (per wave; barrier = all)
        __syncthreads();
        if (t + 1 < 72) stage(t + 1, buf ^ 1);   // overlap DMA with WMMA

        const int am = et * 16 + m16;
        uint4 a0 = *(const uint4*)(&Al[buf][am * 40 + (hi ? 8 : 0)]);
        uint4 a1 = *(const uint4*)(&Al[buf][am * 40 + 16 + (hi ? 8 : 0)]);
        v16bf av = make_frag(a0, a1);
#pragma unroll
        for (int p = 0; p < 2; ++p) {
            const int bn = (np + p) * 16 + m16;
            uint4 b0 = *(const uint4*)(&Bl[buf][bn * 40 + (hi ? 8 : 0)]);
            uint4 b1 = *(const uint4*)(&Bl[buf][bn * 40 + 16 + (hi ? 8 : 0)]);
            acc[p] = wmma_bf16(av, make_frag(b0, b1), acc[p]);
        }
    }

    // --- epilogue: bias + stores
    const float* bias = (conv == 0) ? bq : (conv == 1) ? bk : bv;
    const int eb = e0 + et * 16 + hi * 8;
#pragma unroll
    for (int r = 0; r < 8; ++r) {
        float bsv = bias[eb + r];
        acc[0][r] += bsv;
        acc[1][r] += bsv;
    }

#pragma unroll
    for (int p = 0; p < 2; ++p) {
        const int i = i0 + (np + p) * 16 + m16;
        if (conv < 2) {
            unsigned short* dstT = (conv == 0) ? qT : kT;
            uint4 pk;
            pk.x = (unsigned)f2bf(acc[p][0]) | ((unsigned)f2bf(acc[p][1]) << 16);
            pk.y = (unsigned)f2bf(acc[p][2]) | ((unsigned)f2bf(acc[p][3]) << 16);
            pk.z = (unsigned)f2bf(acc[p][4]) | ((unsigned)f2bf(acc[p][5]) << 16);
            pk.w = (unsigned)f2bf(acc[p][6]) | ((unsigned)f2bf(acc[p][7]) << 16);
            *(uint4*)(dstT + ((size_t)b * NSP + i) * EMB + eb) = pk;
        } else {
#pragma unroll
            for (int r = 0; r < 8; ++r) {
                const size_t off = ((size_t)b * EMB + (eb + r)) * NSP + i;
                vB[off] = f2bf(acc[p][r]);
                vF[off] = acc[p][r];
            }
        }
    }
}

// ---------------------------------------------------------------------------
// Kernel 3: flash attention.  Grid (N/16, B), block 256 (8 waves).
// Waves partition j for S = QK^T and partition e for O += P*V^T; online
// softmax with cross-wave LDS reduction; P shared via LDS bf16.
// ---------------------------------------------------------------------------
__global__ __launch_bounds__(256)
void flash_attn_wmma(const unsigned short* __restrict__ qT,
                     const unsigned short* __restrict__ kT,
                     const unsigned short* __restrict__ vB,
                     const float* __restrict__ vF,
                     float* __restrict__ out) {
    __shared__ unsigned short Ql[16 * 520];   // Q tile 16x512 bf16, padded
    __shared__ unsigned short Pl[16 * 136];   // P tile 16x128 bf16, padded
    __shared__ float redmax[8][16];
    __shared__ float redsum[8][16];

    const int tid  = threadIdx.x;
    const int wave = tid >> 5;
    const int lane = tid & 31;
    const int m16  = lane & 15;
    const int hi   = lane >> 4;

    const int b  = blockIdx.y;
    const int i0 = blockIdx.x * 16;

    // load Q tile (16 rows x 512 bf16) via async b128
    {
        const int row = tid >> 4;
        const int cs  = (tid & 15) * 32;
        const unsigned short* src = qT + ((size_t)b * NSP + i0 + row) * EMB + cs;
        unsigned short* dst = &Ql[row * 520 + cs];
#pragma unroll
        for (int u = 0; u < 4; ++u)
            async_copy_b128(src + u * 8, dst + u * 8);
    }
    wait_async();
    __syncthreads();

    v8f zero = {};
    v8f acc[4];
#pragma unroll
    for (int t = 0; t < 4; ++t) acc[t] = zero;
    float mrun[8], lrun[8];
#pragma unroll
    for (int r = 0; r < 8; ++r) { mrun[r] = -3.0e38f; lrun[r] = 0.f; }

    for (int jb = 0; jb < NSP; jb += 128) {
        // ---- S = Q K^T for this wave's 16 j columns
        v8f s = zero;
        const int j = jb + wave * 16 + m16;
        const unsigned short* krow = kT + ((size_t)b * NSP + j) * EMB;
#pragma unroll
        for (int kc = 0; kc < 16; ++kc) {
            uint4 a0 = *(const uint4*)(&Ql[m16 * 520 + kc * 32 + (hi ? 8 : 0)]);
            uint4 a1 = *(const uint4*)(&Ql[m16 * 520 + kc * 32 + 16 + (hi ? 8 : 0)]);
            uint4 b0 = *(const uint4*)(krow + kc * 32 + (hi ? 8 : 0));
            uint4 b1 = *(const uint4*)(krow + kc * 32 + 16 + (hi ? 8 : 0));
            s = wmma_bf16(make_frag(a0, a1), make_frag(b0, b1), s);
        }

        // ---- per-row max within wave
        float mw[8];
#pragma unroll
        for (int r = 0; r < 8; ++r) {
            float v = s[r];
            for (int msk = 1; msk < 16; msk <<= 1) v = fmaxf(v, __shfl_xor(v, msk));
            mw[r] = v;
        }
        if (m16 == 0) {
#pragma unroll
            for (int r = 0; r < 8; ++r) redmax[wave][hi * 8 + r] = mw[r];
        }
        __syncthreads();

        float mnew[8];
#pragma unroll
        for (int r = 0; r < 8; ++r) {
            float m = mrun[r];
#pragma unroll
            for (int w = 0; w < 8; ++w) m = fmaxf(m, redmax[w][hi * 8 + r]);
            mnew[r] = m;
        }
#pragma unroll
        for (int r = 0; r < 8; ++r) {
            float sc = __expf(mrun[r] - mnew[r]);
            lrun[r] *= sc;
#pragma unroll
            for (int t = 0; t < 4; ++t) acc[t][r] *= sc;
            mrun[r] = mnew[r];
        }

        // ---- P = exp(S - m), row sums, publish P (bf16) to LDS
        float sw[8];
#pragma unroll
        for (int r = 0; r < 8; ++r) {
            float p = __expf(s[r] - mnew[r]);
            s[r] = p;
            float v = p;
            for (int msk = 1; msk < 16; msk <<= 1) v += __shfl_xor(v, msk);
            sw[r] = v;
        }
        if (m16 == 0) {
#pragma unroll
            for (int r = 0; r < 8; ++r) redsum[wave][hi * 8 + r] = sw[r];
        }
#pragma unroll
        for (int r = 0; r < 8; ++r)
            Pl[(hi * 8 + r) * 136 + wave * 16 + m16] = f2bf(s[r]);
        __syncthreads();

#pragma unroll
        for (int r = 0; r < 8; ++r) {
            float l = 0.f;
#pragma unroll
            for (int w = 0; w < 8; ++w) l += redsum[w][hi * 8 + r];
            lrun[r] += l;
        }

        // ---- O += P (16x128) * V^T (128x64) for this wave's e-range
        const int e0w = wave * 64;
#pragma unroll
        for (int kc = 0; kc < 4; ++kc) {
            uint4 a0 = *(const uint4*)(&Pl[m16 * 136 + kc * 32 + (hi ? 8 : 0)]);
            uint4 a1 = *(const uint4*)(&Pl[m16 * 136 + kc * 32 + 16 + (hi ? 8 : 0)]);
            v16bf pa = make_frag(a0, a1);
#pragma unroll
            for (int etl = 0; etl < 4; ++etl) {
                const int e = e0w + etl * 16 + m16;
                const unsigned short* vrow =
                    vB + ((size_t)b * EMB + e) * NSP + jb + kc * 32;
                uint4 b0 = *(const uint4*)(vrow + (hi ? 8 : 0));
                uint4 b1 = *(const uint4*)(vrow + 16 + (hi ? 8 : 0));
                acc[etl] = wmma_bf16(pa, make_frag(b0, b1), acc[etl]);
            }
        }
        __syncthreads();
    }

    // ---- epilogue: out[b][e][i] = O/l + v
#pragma unroll
    for (int etl = 0; etl < 4; ++etl) {
        const int e = wave * 64 + etl * 16 + m16;
#pragma unroll
        for (int r = 0; r < 8; ++r) {
            const int i = i0 + hi * 8 + r;
            const size_t off = ((size_t)b * EMB + e) * NSP + i;
            out[off] = acc[etl][r] / lrun[r] + vF[off];
        }
    }
}

// ---------------------------------------------------------------------------
extern "C" void kernel_launch(void* const* d_in, const int* in_sizes, int n_in,
                              void* d_out, int out_size, void* d_ws, size_t ws_size,
                              hipStream_t stream) {
    (void)in_sizes; (void)n_in; (void)out_size; (void)ws_size;

    const float* feat = (const float*)d_in[0];
    const float* Wq   = (const float*)d_in[1];
    const float* bq   = (const float*)d_in[2];
    const float* Wk   = (const float*)d_in[3];
    const float* bk   = (const float*)d_in[4];
    const float* Wv   = (const float*)d_in[5];
    const float* bv   = (const float*)d_in[6];
    float* out = (float*)d_out;

    char* ws = (char*)d_ws;
    const size_t SZ_BF = (size_t)8 * NSP * EMB * 2;            // 18,874,368
    unsigned short* qT    = (unsigned short*)(ws);
    unsigned short* kT    = (unsigned short*)(ws + SZ_BF);
    unsigned short* vB    = (unsigned short*)(ws + 2 * SZ_BF);
    float*          vF    = (float*)         (ws + 3 * SZ_BF);  // 2*SZ_BF bytes
    unsigned short* Wmat  = (unsigned short*)(ws + 5 * SZ_BF);  // 7,077,888
    unsigned short* featT = (unsigned short*)(ws + 5 * SZ_BF + (size_t)3 * EMB * KCONV * 2);

    repack_w_kernel<<<dim3((3 * EMB * KCONV) / 256), dim3(256), 0, stream>>>(
        Wq, Wk, Wv, Wmat);
    featT_kernel<<<dim3((8 * HW * HW * CIN) / 256), dim3(256), 0, stream>>>(
        feat, featT);
    conv_qkv_wmma<<<dim3(NSP / 64, EMB / 64, 8 * 3), dim3(256), 0, stream>>>(
        featT, Wmat, bq, bk, bv, qT, kT, vB, vF);
    flash_attn_wmma<<<dim3(NSP / 16, 8), dim3(256), 0, stream>>>(
        qT, kT, vB, vF, out);
}
